// MultiLayerFastLocalGraphModelV2_17841294148338
// MI455X (gfx1250) — matrix operand order repaired and checked by reading, not compiled, not run on hardware
//
#include <hip/hip_runtime.h>
#include <hip/hip_bf16.h>
#include <cstdint>

// ---------------------------------------------------------------------------
// MI455X (gfx1250) Point-GNN forward pass.
// GEMMs: v_wmma_f32_16x16x32_f16, double-buffered LDS tiles staged with
// GLOBAL_LOAD_ASYNC_TO_LDS_B128 (ASYNCcnt) when the toolchain exposes it.
// ---------------------------------------------------------------------------

typedef __attribute__((ext_vector_type(16))) _Float16 v16h;
typedef __attribute__((ext_vector_type(8)))  float    v8f;
typedef _Float16 half_t;

#define EPS_BN 1e-5f

#if __has_builtin(__builtin_amdgcn_global_load_async_to_lds_b128)
#define ASYNC_LDS 1
#else
#define ASYNC_LDS 0
#endif

#if __has_builtin(__builtin_amdgcn_s_wait_asynccnt)
#define WAIT_ASYNC(n) __builtin_amdgcn_s_wait_asynccnt(n)
#elif ASYNC_LDS
#define WAIT_ASYNC(n) asm volatile("s_wait_asynccnt %0" ::"i"(n) : "memory")
#else
#define WAIT_ASYNC(n) ((void)0)
#endif

#if ASYNC_LDS
typedef int v4i_gcc __attribute__((__vector_size__(16)));
typedef __attribute__((address_space(1))) v4i_gcc* as1_v4i_p;
typedef __attribute__((address_space(3))) v4i_gcc* as3_v4i_p;
#endif

__device__ __forceinline__ void copy16(const half_t* g, half_t* l)
{
#if ASYNC_LDS
    // AS1 pointers are bit-identical to generic global pointers; AS3 pointers
    // are the low 32 bits of a generic LDS pointer (the LDS offset).
    __builtin_amdgcn_global_load_async_to_lds_b128(
        (as1_v4i_p)(unsigned long long)(uintptr_t)g,
        (as3_v4i_p)(unsigned)(uintptr_t)l, 0, 0);
#else
    *(uint4*)l = *(const uint4*)g;
#endif
}

// ====================== WMMA GEMM: C = relu(A @ W^T + b) =====================
// A: [M x Kp] f16 (lda = Kp, mult of 32, zero padded)
// W: [Npad64 x Kp] f16 (row n = output feature n, zero padded rows to 64)
// C: [M x ldc] f32. Block = 256 thr = 8 waves; tile = 128 rows x 64 cols.
__global__ __launch_bounds__(256)
void gemm_bias_relu(const half_t* __restrict__ A, int lda,
                    const half_t* __restrict__ W, int ldw,
                    const float* __restrict__ bias,
                    float* __restrict__ C, int ldc,
                    int M, int N, int Kp)
{
    __shared__ __align__(16) half_t sA[2][128 * 32];
    __shared__ __align__(16) half_t sB[2][64 * 32];

    const int tid    = threadIdx.x;
    const int lane   = tid & 31;
    const int wave   = tid >> 5;
    const int blockM = blockIdx.x * 128;
    const int colB   = blockIdx.y * 64;
    const int m      = lane & 15;
    const int hf     = lane >> 4;

    // Per-thread staging slots: uniform 3x b128 per stage per thread.
    const int arow0 = tid >> 2;             // A rows 0..63
    const int ac    = (tid & 3) << 3;       // half offset within row
    const int arow1 = arow0 + 64;           // A rows 64..127
    const int brow  = tid >> 2;             // B rows (cols of C) 0..63

    v8f acc[4] = {{}, {}, {}, {}};

    auto stage = [&](int b, int kk) {
        int gr0 = blockM + arow0; if (gr0 >= M) gr0 = M - 1;   // clamp: pad rows
        int gr1 = blockM + arow1; if (gr1 >= M) gr1 = M - 1;   // are discarded
        copy16(A + (size_t)gr0 * lda + kk + ac, &sA[b][arow0 * 32 + ac]);
        copy16(A + (size_t)gr1 * lda + kk + ac, &sA[b][arow1 * 32 + ac]);
        copy16(W + (size_t)(colB + brow) * ldw + kk + ac, &sB[b][brow * 32 + ac]);
    };

    stage(0, 0);
    int buf = 0;
    for (int kk = 0; kk < Kp; kk += 32) {
        if (kk + 32 < Kp) { stage(buf ^ 1, kk + 32); WAIT_ASYNC(3); }
        else              { WAIT_ASYNC(0); }
        __syncthreads();

        // repack to CDNA5 WMMA VGPR layouts (16-bit A 16x32 / B 32x16)
        union { v16h v; unsigned u[8]; } a, b;
        const int rowL = wave * 16 + m;
#pragma unroll
        for (int p = 0; p < 8; ++p) {
            int kidx = 16 * hf + 2 * p;
            a.u[p] = *(const unsigned*)(&sA[buf][rowL * 32 + kidx]);
        }
#pragma unroll
        for (int t = 0; t < 4; ++t) {
#pragma unroll
            for (int p = 0; p < 8; ++p)
                b.u[p] = *(const unsigned*)(&sB[buf][(16 * t + m) * 32 + 16 * hf + 2 * p]);
            acc[t] = __builtin_amdgcn_wmma_f32_16x16x32_f16(
                false, a.v, false, b.v, (short)0, acc[t], false, false);
        }
        __syncthreads();
        buf ^= 1;
    }

    // epilogue: lane = N, vgpr r -> M = r + 8*hf
#pragma unroll
    for (int t = 0; t < 4; ++t) {
        int gcol = colB + 16 * t + m;
        if (gcol < N) {
            float bv = bias[gcol];
#pragma unroll
            for (int r = 0; r < 8; ++r) {
                int gm = blockM + wave * 16 + r + 8 * hf;
                if (gm < M)
                    C[(size_t)gm * ldc + gcol] = fmaxf(acc[t][r] + bv, 0.f);
            }
        }
    }
}

// ====================== Batch-norm (batch statistics) ========================
__global__ __launch_bounds__(256)
void col_stats(const float* __restrict__ X, int R, int ld,
               float* __restrict__ mean, float* __restrict__ var)
{
    __shared__ float ss[256], sq[256];
    int c = blockIdx.x;
    float s = 0.f, q = 0.f;
    for (int r = threadIdx.x; r < R; r += 256) {
        float v = X[(size_t)r * ld + c];
        s += v; q += v * v;
    }
    ss[threadIdx.x] = s; sq[threadIdx.x] = q;
    __syncthreads();
    for (int st = 128; st > 0; st >>= 1) {
        if (threadIdx.x < st) {
            ss[threadIdx.x] += ss[threadIdx.x + st];
            sq[threadIdx.x] += sq[threadIdx.x + st];
        }
        __syncthreads();
    }
    if (threadIdx.x == 0) {
        float mu = ss[0] / (float)R;
        mean[c] = mu;
        var[c]  = sq[0] / (float)R - mu * mu;
    }
}

__global__ void bn_apply(float* X, int R, int C, int ld,
                         const float* mean, const float* var,
                         const float* gamma, const float* beta)
{
    long long i = (long long)blockIdx.x * blockDim.x + threadIdx.x;
    if (i >= (long long)R * C) return;
    int r = (int)(i / C), c = (int)(i % C);
    float v = X[(size_t)r * ld + c];
    X[(size_t)r * ld + c] =
        gamma[c] * (v - mean[c]) * rsqrtf(var[c] + EPS_BN) + beta[c];
}

// ====================== Conversions / elementwise ============================
__global__ void to_half_k(const float* __restrict__ X, int R, int C, int ldin,
                          half_t* __restrict__ H, int ldout)
{
    long long i = (long long)blockIdx.x * blockDim.x + threadIdx.x;
    if (i >= (long long)R * ldout) return;
    int c = (int)(i % ldout);
    int r = (int)(i / ldout);
    H[i] = (c < C) ? (half_t)X[(size_t)r * ldin + c] : (half_t)0.f;
}

__global__ void wconv(const float* __restrict__ W, int dout, int din,
                      half_t* __restrict__ Wh, int Kpad, long long n)
{
    long long i = (long long)blockIdx.x * blockDim.x + threadIdx.x;
    if (i >= n) return;
    int k = (int)(i % Kpad);
    int r = (int)(i / Kpad);
    Wh[i] = (r < dout && k < din) ? (half_t)W[(size_t)r * din + k] : (half_t)0.f;
}

__global__ void zero_f32(float* p, long long n)
{
    long long i = (long long)blockIdx.x * blockDim.x + threadIdx.x;
    if (i < n) p[i] = 0.f;
}

__global__ void add_rows(float* Y, const float* X, int R, int C, int ld)
{
    long long i = (long long)blockIdx.x * blockDim.x + threadIdx.x;
    if (i >= (long long)R * C) return;
    int r = (int)(i / C), c = (int)(i % C);
    Y[(size_t)r * ld + c] += X[(size_t)r * ld + c];
}

__global__ void mask_rows(const float* X, const float* keep, float* Y,
                          int R, int C, int ld)
{
    long long i = (long long)blockIdx.x * blockDim.x + threadIdx.x;
    if (i >= (long long)R * C) return;
    int r = (int)(i / C), c = (int)(i % C);
    Y[(size_t)r * ld + c] = X[(size_t)r * ld + c] * keep[r];
}

// ====================== Gathers / scatters ===================================
__global__ void gather_feats(const float* __restrict__ pf,
                             const float* __restrict__ pc,
                             const int* __restrict__ kpi,
                             const int* __restrict__ si,
                             int S, half_t* __restrict__ H)
{
    long long i = (long long)blockIdx.x * blockDim.x + threadIdx.x;
    if (i >= (long long)S * 32) return;
    int s = (int)(i / 32), c = (int)(i % 32);
    float v = 0.f;
    int p  = si[2 * s];
    int kq = si[2 * s + 1];
    if (c == 0)      v = pf[p];
    else if (c < 4) {
        int kp = kpi[kq];
        v = pc[(size_t)p * 3 + (c - 1)] - pc[(size_t)kp * 3 + (c - 1)];
    }
    H[i] = (half_t)v;
}

__global__ void gather_kc(const float* __restrict__ pc,
                          const int* __restrict__ kpi,
                          float* __restrict__ kc, int K)
{
    long long i = (long long)blockIdx.x * blockDim.x + threadIdx.x;
    if (i >= (long long)K * 3) return;
    int r = (int)(i / 3), c = (int)(i % 3);
    kc[i] = pc[(size_t)kpi[r] * 3 + c];
}

__global__ void add_coords(const float* kc, const float* delta, int ldd,
                           float* vcoff, int K)
{
    long long i = (long long)blockIdx.x * blockDim.x + threadIdx.x;
    if (i >= (long long)K * 3) return;
    int r = (int)(i / 3), c = (int)(i % 3);
    vcoff[i] = kc[i] + delta[(size_t)r * ldd + c];
}

__global__ void gather_edge(const float* __restrict__ vf, int ldv,
                            const float* __restrict__ kc,
                            const float* __restrict__ vcoff,
                            const int* __restrict__ ed,
                            int E, half_t* __restrict__ H, int ldh)
{
    long long i = (long long)blockIdx.x * blockDim.x + threadIdx.x;
    if (i >= (long long)E * ldh) return;
    int e = (int)(i / ldh), c = (int)(i % ldh);
    half_t v = (half_t)0.f;
    int e0 = ed[2 * e], e1 = ed[2 * e + 1];
    if (c < 300)       v = (half_t)vf[(size_t)e0 * ldv + c];
    else if (c < 303) {
        int d = c - 300;
        v = (half_t)(kc[(size_t)e0 * 3 + d] - vcoff[(size_t)e1 * 3 + d]);
    }
    H[i] = v;
}

// segment-max with relu clamp; Out pre-zeroed; values >= 0 so int-pattern
// atomicMax is order-correct and reproduces max(segment_max, 0).
__global__ void scatter_max_rows(const float* __restrict__ X, int R, int C, int ld,
                                 const int* __restrict__ idx, int stride, int off,
                                 float* __restrict__ Out, int ldo)
{
    long long i = (long long)blockIdx.x * blockDim.x + threadIdx.x;
    if (i >= (long long)R * C) return;
    int r = (int)(i / C), c = (int)(i % C);
    float v = fmaxf(X[(size_t)r * ld + c], 0.f);
    int s = idx[(size_t)r * stride + off];
    atomicMax((int*)&Out[(size_t)s * ldo + c], __float_as_int(v));
}

// ====================== Gate / top-k =========================================
__global__ void gate_logits(const float* __restrict__ H, int ldh,
                            const float* __restrict__ W2,
                            const float* __restrict__ b2,
                            const float* __restrict__ noise,
                            float* __restrict__ scores, int K)
{
    int i = blockIdx.x * blockDim.x + threadIdx.x;
    if (i >= K) return;
    float a = b2[0];
    for (int j = 0; j < 75; ++j) a += H[(size_t)i * ldh + j] * W2[j];
    scores[i] = 1.f / (1.f + expf(-a)) + noise[i];   // softmax is monotone
}

__global__ void topk_mask(const float* __restrict__ s, int K, int k,
                          float* __restrict__ keep)
{
    int i = blockIdx.x * blockDim.x + threadIdx.x;
    if (i >= K) return;
    float si = s[i];
    int cnt = 0;
    for (int j = 0; j < K; ++j) {
        float sj = s[j];
        cnt += (sj > si) || (sj == si && j < i);
    }
    keep[i] = (cnt < k) ? 1.f : 0.f;
}

// ====================== Small heads (no relu) ================================
__global__ void head_cls(const float* __restrict__ X, int ldx,
                         const float* __restrict__ Wf,
                         const float* __restrict__ bf,
                         float* __restrict__ out, int K)
{
    long long i = (long long)blockIdx.x * blockDim.x + threadIdx.x;
    if (i >= (long long)K * 4) return;
    int r = (int)(i / 4), n = (int)(i % 4);
    float a = bf[n];
    for (int j = 0; j < 64; ++j) a += X[(size_t)r * ldx + j] * Wf[n * 64 + j];
    out[i] = a;
}

__global__ void head_loc(const float* __restrict__ X, int ldx,
                         const float* __restrict__ Wf,
                         const float* __restrict__ bf,
                         float* __restrict__ out, int K, int h)
{
    long long i = (long long)blockIdx.x * blockDim.x + threadIdx.x;
    if (i >= (long long)K * 7) return;
    int r = (int)(i / 7), n = (int)(i % 7);
    float a = bf[n];
    for (int j = 0; j < 64; ++j) a += X[(size_t)r * ldx + j] * Wf[n * 64 + j];
    out[(size_t)(r * 4 + h) * 7 + n] = a;
}

// ============================================================================
extern "C" void kernel_launch(void* const* d_in, const int* in_sizes, int n_in,
                              void* d_out, int out_size, void* d_ws, size_t ws_size,
                              hipStream_t stream)
{
    (void)in_sizes; (void)n_in; (void)out_size; (void)ws_size;
    const int K = 10000, S = 160000, E = 160000;

    const float* pf  = (const float*)d_in[0];
    const float* pc  = (const float*)d_in[1];
    const int*   kpi = (const int*)d_in[2];
    const int*   si  = (const int*)d_in[3];
    const int*   ed  = (const int*)d_in[4];
    const float* gnz = (const float*)d_in[5];
    float* out = (float*)d_out;

    auto cd  = [](long long n) { return (unsigned)((n + 255) / 256); };
    auto a16 = [](int x) { return (x + 15) & ~15; };
    auto a32 = [](int x) { return (x + 31) & ~31; };
    auto a64 = [](int x) { return (x + 63) & ~63; };

    // ---- workspace arena ----
    char* base = (char*)d_ws;
    size_t off = 0;
    auto alloc = [&](size_t bytes) -> void* {
        void* p = base + off;
        off += (bytes + 255) & ~(size_t)255;
        return p;
    };
    float*  bufA = (float*)alloc((size_t)S * 304 * 4);
    float*  bufB = (float*)alloc((size_t)S * 304 * 4);
    half_t* bufH = (half_t*)alloc((size_t)S * 320 * 2);
    float*  kf   = (float*)alloc((size_t)K * 304 * 4);
    float*  kb1  = (float*)alloc((size_t)K * 304 * 4);
    float*  kb2  = (float*)alloc((size_t)K * 304 * 4);
    float*  kb3  = (float*)alloc((size_t)K * 304 * 4);
    half_t* kH   = (half_t*)alloc((size_t)K * 320 * 2);
    float*  kc   = (float*)alloc((size_t)K * 3 * 4);
    float*  vco  = (float*)alloc((size_t)K * 3 * 4);
    float*  scores = (float*)alloc((size_t)K * 4);
    float*  keep   = (float*)alloc((size_t)K * 4);
    float*  meanB  = (float*)alloc(304 * 4);
    float*  varB   = (float*)alloc(304 * 4);

    // ---- weight prep (fp32 -> padded f16, rows padded to 64) ----
    struct WH { half_t* Wh; const float* b; const float* beta; const float* g;
                int dout, din, Kpad, ldc; };
    auto prep = [&](int wi, int bi, int betai, int gi, int dout, int din) -> WH {
        WH w;
        w.dout = dout; w.din = din; w.Kpad = a32(din); w.ldc = a16(dout);
        w.b = (const float*)d_in[bi];
        w.beta = betai >= 0 ? (const float*)d_in[betai] : nullptr;
        w.g    = gi    >= 0 ? (const float*)d_in[gi]    : nullptr;
        long long n = (long long)a64(dout) * w.Kpad;
        w.Wh = (half_t*)alloc((size_t)n * 2);
        wconv<<<cd(n), 256, 0, stream>>>((const float*)d_in[wi], dout, din,
                                         w.Wh, w.Kpad, n);
        return w;
    };
    auto gemm = [&](const half_t* H, const WH& w, float* C, int M) {
        dim3 g((M + 127) / 128, (w.dout + 63) / 64);
        gemm_bias_relu<<<g, 256, 0, stream>>>(H, w.Kpad, w.Wh, w.Kpad, w.b,
                                              C, w.ldc, M, w.dout, w.Kpad);
    };
    auto bn = [&](float* X, int R, const WH& w) {
        col_stats<<<w.dout, 256, 0, stream>>>(X, R, w.ldc, meanB, varB);
        bn_apply<<<cd((long long)R * w.dout), 256, 0, stream>>>(
            X, R, w.dout, w.ldc, meanB, varB, w.g, w.beta);
    };
    auto toH = [&](const float* X, int R, int C, int ldin, half_t* H, int ldout) {
        to_half_k<<<cd((long long)R * ldout), 256, 0, stream>>>(X, R, C, ldin, H, ldout);
    };
    auto zero = [&](float* p, long long n) {
        zero_f32<<<cd(n), 256, 0, stream>>>(p, n);
    };

    // params flattened in jax pytree (sorted-key) order:
    // cls(6..11), gnn[i] at 12+28i, loc[h] at 96+10h, psp_out 136.., psp_point 144..
    WH pp0 = prep(144, 145, 146, 147,  32,   4);
    WH pp1 = prep(148, 149, 150, 151,  64,  32);
    WH pp2 = prep(152, 153, 154, 155, 128,  64);
    WH pp3 = prep(156, 157, 158, 159, 300, 128);
    WH po0 = prep(136, 137, 138, 139, 300, 300);
    WH po1 = prep(140, 141, 142, 143, 300, 300);
    WH aut0[3], aut1[3], edg0[3], edg1[3], gw1[3], upd0[3], upd1[3];
    for (int i = 0; i < 3; ++i) {
        int g = 12 + 28 * i;
        aut0[i] = prep(g + 0,  g + 1,  g + 2,  g + 3,   64, 300);
        aut1[i] = prep(g + 4,  g + 5,  g + 6,  g + 7,    3,  64);
        edg0[i] = prep(g + 8,  g + 9,  g + 10, g + 11, 300, 303);
        edg1[i] = prep(g + 12, g + 13, g + 14, g + 15, 300, 300);
        gw1[i]  = prep(g + 16, g + 18, -1, -1,          75, 300);
        upd0[i] = prep(g + 20, g + 21, g + 22, g + 23, 300, 300);
        upd1[i] = prep(g + 24, g + 25, g + 26, g + 27, 300, 300);
    }
    WH clsm = prep(8, 9, 10, 11, 64, 300);
    WH locm0[4], locm1[4];
    for (int h = 0; h < 4; ++h) {
        int b = 96 + 10 * h;
        locm0[h] = prep(b + 2, b + 3, b + 4, b + 5, 300, 300);
        locm1[h] = prep(b + 6, b + 7, b + 8, b + 9,  64, 300);
    }

    // ================= pipeline =================
    // psp_point on S rows
    gather_feats<<<cd((long long)S * 32), 256, 0, stream>>>(pf, pc, kpi, si, S, bufH);
    gemm(bufH, pp0, bufA, S); bn(bufA, S, pp0); toH(bufA, S,  32,  32, bufH,  32);
    gemm(bufH, pp1, bufA, S); bn(bufA, S, pp1); toH(bufA, S,  64,  64, bufH,  64);
    gemm(bufH, pp2, bufA, S); bn(bufA, S, pp2); toH(bufA, S, 128, 128, bufH, 128);
    gemm(bufH, pp3, bufA, S); bn(bufA, S, pp3);

    // segment-max S -> K
    zero(kb1, (long long)K * 304);
    scatter_max_rows<<<cd((long long)S * 300), 256, 0, stream>>>(
        bufA, S, 300, 304, si, 2, 1, kb1, 304);

    // psp_out on K rows -> kf
    toH(kb1, K, 300, 304, kH, 320);
    gemm(kH, po0, kb2, K); bn(kb2, K, po0); toH(kb2, K, 300, 304, kH, 320);
    gemm(kH, po1, kf, K);  bn(kf,  K, po1);

    gather_kc<<<cd((long long)K * 3), 256, 0, stream>>>(pc, kpi, kc, K);

    // GNN layers
    for (int i = 0; i < 3; ++i) {
        int g = 12 + 28 * i;
        // auto MLP -> coordinate offsets
        toH(kf, K, 300, 304, kH, 320);
        gemm(kH, aut0[i], kb1, K); bn(kb1, K, aut0[i]);
        toH(kb1, K, 64, 64, kH, 64);
        gemm(kH, aut1[i], kb2, K); bn(kb2, K, aut1[i]);
        add_coords<<<cd((long long)K * 3), 256, 0, stream>>>(kc, kb2, 16, vco, K);
        // edge MLP on E rows
        gather_edge<<<cd((long long)E * 320), 256, 0, stream>>>(
            kf, 304, kc, vco, ed, E, bufH, 320);
        gemm(bufH, edg0[i], bufA, E); bn(bufA, E, edg0[i]);
        toH(bufA, E, 300, 304, bufH, 320);
        gemm(bufH, edg1[i], bufB, E); bn(bufB, E, edg1[i]);
        // aggregate
        zero(kb1, (long long)K * 304);
        scatter_max_rows<<<cd((long long)E * 300), 256, 0, stream>>>(
            bufB, E, 300, 304, ed, 2, 1, kb1, 304);
        // update MLP + residual
        toH(kb1, K, 300, 304, kH, 320);
        gemm(kH, upd0[i], kb2, K); bn(kb2, K, upd0[i]);
        toH(kb2, K, 300, 304, kH, 320);
        gemm(kH, upd1[i], kb3, K); bn(kb3, K, upd1[i]);
        add_rows<<<cd((long long)K * 300), 256, 0, stream>>>(kb3, kf, K, 300, 304);
        // gate (forward value == hard top-k mask)
        toH(kb3, K, 300, 304, kH, 320);
        gemm(kH, gw1[i], kb2, K);
        gate_logits<<<cd(K), 256, 0, stream>>>(
            kb2, 80, (const float*)d_in[g + 17], (const float*)d_in[g + 19],
            gnz + (size_t)i * K, scores, K);
        int kv = (i == 0) ? 300 : 30;
        topk_mask<<<cd(K), 256, 0, stream>>>(scores, K, kv, keep);
        mask_rows<<<cd((long long)K * 300), 256, 0, stream>>>(
            kb3, keep, kf, K, 300, 304);
    }

    // cls head -> logits (K x 4)
    toH(kf, K, 300, 304, kH, 320);
    gemm(kH, clsm, kb1, K); bn(kb1, K, clsm);
    head_cls<<<cd((long long)K * 4), 256, 0, stream>>>(
        kb1, 64, (const float*)d_in[6], (const float*)d_in[7], out, K);

    // loc heads -> box (K x 4 x 7)
    for (int h = 0; h < 4; ++h) {
        int b = 96 + 10 * h;
        toH(kf, K, 300, 304, kH, 320);
        gemm(kH, locm0[h], kb1, K); bn(kb1, K, locm0[h]);
        toH(kb1, K, 300, 304, kH, 320);
        gemm(kH, locm1[h], kb2, K); bn(kb2, K, locm1[h]);
        head_loc<<<cd((long long)K * 7), 256, 0, stream>>>(
            kb2, 64, (const float*)d_in[b + 0], (const float*)d_in[b + 1],
            out + (size_t)K * 4, K, h);
    }
}
